// QNetworks_63763084476799
// MI455X (gfx1250) — matrix operand order repaired and checked by reading, not compiled, not run on hardware
//
#include <hip/hip_runtime.h>
#include <hip/hip_bf16.h>
#include <math.h>

// Problem constants (from reference)
#define BB   256
#define PP   512
#define DD   16
#define HH   256
#define AA   8
#define CC   2
#define HID  256
#define INML 265   // H + A + 1

typedef __bf16 bf16x16 __attribute__((ext_vector_type(16)));
typedef float  f32x8   __attribute__((ext_vector_type(8)));

struct alignas(16) B128 { unsigned int w[4]; };
struct alignas(16) B256 { B128 lo; B128 hi; };

static __device__ __forceinline__ unsigned short f32_to_bf16_bits(float f) {
    unsigned u = __builtin_bit_cast(unsigned, f);
    unsigned r = u + 0x7FFFu + ((u >> 16) & 1u);   // round-to-nearest-even
    return (unsigned short)(r >> 16);
}
static __device__ __forceinline__ __bf16 f2bf(float f) {
    return __builtin_bit_cast(__bf16, f32_to_bf16_bits(f));
}
static __device__ __forceinline__ float bf2f(unsigned short b) {
    return __builtin_bit_cast(float, ((unsigned)b) << 16);
}
static __device__ __forceinline__ f32x8 zero8() {
    f32x8 v;
#pragma unroll
    for (int e = 0; e < 8; ++e) v[e] = 0.0f;
    return v;
}
static __device__ __forceinline__ f32x8 wmma_bf16(bf16x16 a, bf16x16 b, f32x8 c) {
    // D = A(16x32 bf16) * B(32x16 bf16) + C(16x16 f32)
    return __builtin_amdgcn_wmma_f32_16x16x32_bf16(false, a, false, b, (short)0, c,
                                                   false, false);
}

// Load 32 contiguous bytes (16 bf16) from LDS as a fragment.
static __device__ __forceinline__ bf16x16 load_frag32B(const unsigned short* p) {
    B256 t;
    t.lo = *(const B128*)(p);
    t.hi = *(const B128*)(p + 8);
    return __builtin_bit_cast(bf16x16, t);
}

// Build a 16-bit A-fragment (16x32) from LDS (row-major [16][rowStride] bf16 bits).
// Contiguous K order (B-frags use the same): lanes 0-15 -> K=[0..15], lanes 16-31 -> K=[16..31]; M = lane%16.
static __device__ __forceinline__ bf16x16 load_a_frag(const unsigned short* base,
                                                      int rowStride, int k0, int lane) {
    int m  = lane & 15;
    int kg = (lane & 16) ? 16 : 0;
    const unsigned short* p = base + m * rowStride + k0 + kg;
    B256 t;
    t.lo = *(const B128*)(p);        // elements 0..7  : K = k0+kg .. +7
    t.hi = *(const B128*)(p + 8);    // elements 8..15 : K = k0+kg+8 .. +15
    return __builtin_bit_cast(bf16x16, t);
}

// LDS layout (byte offsets; MLP region aliases the Wh_n fragment region):
//  GRU : hbuf [2][16][256] bf16 @0      (16384B)
//        xbuf [2][16][32]  bf16 @16384  (2048B)
//        wn   pre-swizzled Wh_n B-frags @18432 (16 colgroups x 8 chunks x 1024B = 131072B)
//  MLP : hid [16][272] f32 @18432 (17408B), h1 [16][512] bf16 @35840 (16384B),
//        h2 [16][512] bf16 @52224 (16384B)  (all dead-alias into wn)
#define HB_OFF  0
#define XB_OFF  16384
#define WN_OFF  18432
#define SMEM_BYTES (18432 + 131072)   // 149504

__global__ __launch_bounds__(512, 1) void qnet_kernel(
    const float* __restrict__ particles, const float* __restrict__ wts,
    const float* __restrict__ actions,   const float* __restrict__ timev,
    const float* __restrict__ Wi,        const float* __restrict__ bi,
    const float* __restrict__ Wh_rz,     const float* __restrict__ Wh_n,
    const float* __restrict__ bn,
    const float* __restrict__ W1, const float* __restrict__ b1,
    const float* __restrict__ W2, const float* __restrict__ b2,
    const float* __restrict__ W3, const float* __restrict__ b3,
    const int* __restrict__ ntsP, float* __restrict__ out)
{
    __shared__ __align__(16) unsigned char smem[SMEM_BYTES];
    unsigned short* hbuf = (unsigned short*)(smem + HB_OFF);  // [2][16][256]
    unsigned short* xbuf = (unsigned short*)(smem + XB_OFF);  // [2][16][32]
    unsigned short* wn   = (unsigned short*)(smem + WN_OFF);  // swizzled Wh_n frags

    const int tid  = threadIdx.x;
    const int lane = tid & 31;
    const int wave = tid >> 5;        // 16 waves
    const int b0   = blockIdx.x * 16; // batch tile

    const int jloc = lane & 15;
    const int jcol = wave * 16 + jloc;        // hidden column owned (C-fragment N)
    const int kbB  = (lane & 16) ? 16 : 0;    // B-fragment K base for this lane half
    const int mofs = (lane & 16) ? 8 : 0;     // C-fragment M offset

    // ---------------- Preload r/z recurrent weights as B-fragments (registers) -----
    // gate 0 = r (Wh_rz col jcol), gate 1 = z (Wh_rz col 256+jcol)
    bf16x16 wB[2][8];
    bf16x16 wI[3];
#pragma unroll
    for (int g = 0; g < 2; ++g) {
        const int col = (g == 1) ? (HH + jcol) : jcol;
#pragma unroll
        for (int kc = 0; kc < 8; ++kc) {
            bf16x16 v;
#pragma unroll
            for (int e = 0; e < 16; ++e) {
                int k = kc * 32 + kbB + e;
                v[e] = f2bf(Wh_rz[k * (2 * HH) + col]);
            }
            wB[g][kc] = v;
        }
    }
#pragma unroll
    for (int g = 0; g < 3; ++g) {
        bf16x16 u;
#pragma unroll
        for (int e = 0; e < 16; ++e) {
            int k = kbB + e;
            float f = (k < DD + 1) ? Wi[k * (3 * HH) + g * HH + jcol] : 0.0f;
            u[e] = f2bf(f);
        }
        wI[g] = u;
    }
    // gate n (Wh_n) -> pre-swizzled B-fragments in LDS (this wave's col-group only)
#pragma unroll 1
    for (int kc = 0; kc < 8; ++kc) {
        unsigned short* dst = wn + (wave * 8 + kc) * 512 + lane * 16;
#pragma unroll
        for (int e = 0; e < 16; ++e) {
            int k = kc * 32 + kbB + e;
            dst[e] = f32_to_bf16_bits(Wh_n[k * HH + jcol]);
        }
    }
    const float biR = bi[jcol];
    const float biZ = bi[HH + jcol];
    const float biN = bi[2 * HH + jcol];
    const float bnv = bn[jcol];

    // ---------------- Init h=0 and stage x(p=0) -------------------------------------
    for (int idx = tid; idx < 16 * 256; idx += 512) hbuf[idx] = 0;  // buffer 0
    {
        int m = tid >> 5, k = tid & 31;
        float f = 0.0f;
        if (k < DD)       f = particles[((size_t)(b0 + m) * PP + 0) * DD + k];
        else if (k == DD) f = wts[(size_t)(b0 + m) * PP + 0];
        xbuf[m * 32 + k] = f32_to_bf16_bits(f);  // buffer 0
    }
    f32x8 hfrag = zero8();
    __syncthreads();

    // ---------------- GRU scan over P=512 particles --------------------------------
    int cur = 0;
#pragma unroll 1
    for (int p = 0; p < PP; ++p) {
        const int nxt = cur ^ 1;
        f32x8 accR = zero8(), accZ = zero8(), accG = zero8(), accI = zero8();

        // x_t @ Wi  (K padded 17 -> 32)
        {
            bf16x16 xA = load_a_frag(xbuf + cur * 16 * 32, 32, 0, lane);
            accR = wmma_bf16(xA, wI[0], accR);
            accZ = wmma_bf16(xA, wI[1], accZ);
            accI = wmma_bf16(xA, wI[2], accI);
        }
        // h @ [Wh_r | Wh_z | Wh_n]  (K = 256 in 8 chunks)
#pragma unroll
        for (int kc = 0; kc < 8; ++kc) {
            bf16x16 hA = load_a_frag(hbuf + cur * 16 * 256, 256, kc * 32, lane);
            bf16x16 wN = load_frag32B(wn + (wave * 8 + kc) * 512 + lane * 16);
            accR = wmma_bf16(hA, wB[0][kc], accR);
            accZ = wmma_bf16(hA, wB[1][kc], accZ);
            accG = wmma_bf16(hA, wN, accG);
        }
        // gate math on matching C-fragments (N = jcol for all accumulators)
#pragma unroll
        for (int e = 0; e < 8; ++e) {
            float r  = 1.0f / (1.0f + expf(-(accR[e] + biR)));
            float z  = 1.0f / (1.0f + expf(-(accZ[e] + biZ)));
            float n  = tanhf(accI[e] + biN + r * (accG[e] + bnv));
            float hn = (1.0f - z) * n + z * hfrag[e];
            hfrag[e] = hn;
            int M = e + mofs;
            hbuf[(nxt * 16 + M) * 256 + jcol] = f32_to_bf16_bits(hn);
        }
        // stage x for step p+1 (all 512 threads, one element each)
        {
            int m = tid >> 5, k = tid & 31, pp = p + 1;
            float f = 0.0f;
            if (pp < PP) {
                if (k < DD)       f = particles[((size_t)(b0 + m) * PP + pp) * DD + k];
                else if (k == DD) f = wts[(size_t)(b0 + m) * PP + pp];
            }
            xbuf[(nxt * 16 + m) * 32 + k] = f32_to_bf16_bits(f);
        }
        __syncthreads();
        cur = nxt;
    }

    // ---------------- MLP head (f32 VALU; ~0.25% of FLOPs) -------------------------
    float* hid          = (float*)(smem + WN_OFF);                 // [16][272]
    unsigned short* h1s = (unsigned short*)(smem + WN_OFF + 17408); // [16][512]
    unsigned short* h2s = (unsigned short*)(smem + WN_OFF + 34816 - 1024); // 33792+18432
    // (h1 @ +17408, h2 @ +33792 relative to WN_OFF)

    // enc (f32 from register C-fragments) -> hidden[:, :256]
#pragma unroll
    for (int e = 0; e < 8; ++e) {
        int M = e + mofs;
        hid[M * 272 + jcol] = hfrag[e];
    }
    // actions, normalized time, zero pad -> hidden[:, 256:272]
    if (tid < 16 * 16) {
        int m = tid >> 4, i = tid & 15;
        float v = 0.0f;
        if (i < AA)       v = actions[(b0 + m) * AA + i];
        else if (i == AA) v = timev[b0 + m] / (float)ntsP[0];
        hid[m * 272 + 256 + i] = v;
    }
    __syncthreads();

    // layer 1: [16,265] x [265,256] per critic, relu
#pragma unroll 1
    for (int it = 0; it < 16; ++it) {
        int m = it, c = tid >> 8, j = tid & 255;
        float s = b1[c * HID + j];
#pragma unroll 4
        for (int i = 0; i < INML; ++i)
            s += hid[m * 272 + i] * W1[((size_t)c * INML + i) * HID + j];
        s = fmaxf(s, 0.0f);
        h1s[m * 512 + c * HID + j] = f32_to_bf16_bits(s);
    }
    __syncthreads();

    // layer 2: [16,256] x [256,256] per critic, relu
#pragma unroll 1
    for (int it = 0; it < 16; ++it) {
        int m = it, c = tid >> 8, j = tid & 255;
        float s = b2[c * HID + j];
#pragma unroll 4
        for (int i = 0; i < HID; ++i)
            s += bf2f(h1s[m * 512 + c * HID + i]) * W2[((size_t)c * HID + i) * HID + j];
        s = fmaxf(s, 0.0f);
        h2s[m * 512 + c * HID + j] = f32_to_bf16_bits(s);
    }
    __syncthreads();

    // layer 3: dot with W3[c,:,0] -> q[b, c]
    if (tid < 32) {
        int m = tid >> 1, c = tid & 1;
        float s = b3[c];
#pragma unroll 4
        for (int i = 0; i < HID; ++i)
            s += bf2f(h2s[m * 512 + c * HID + i]) * W3[c * HID + i];
        out[(b0 + m) * CC + c] = s;
    }
}

extern "C" void kernel_launch(void* const* d_in, const int* in_sizes, int n_in,
                              void* d_out, int out_size, void* d_ws, size_t ws_size,
                              hipStream_t stream) {
    (void)in_sizes; (void)n_in; (void)out_size; (void)d_ws; (void)ws_size;
    const float* particles = (const float*)d_in[0];
    const float* weights   = (const float*)d_in[1];
    const float* actions   = (const float*)d_in[2];
    const float* timev     = (const float*)d_in[3];
    const float* Wi        = (const float*)d_in[4];
    const float* bi        = (const float*)d_in[5];
    const float* Wh_rz     = (const float*)d_in[6];
    const float* Wh_n      = (const float*)d_in[7];
    const float* bn        = (const float*)d_in[8];
    const float* W1        = (const float*)d_in[9];
    const float* b1        = (const float*)d_in[10];
    const float* W2        = (const float*)d_in[11];
    const float* b2        = (const float*)d_in[12];
    const float* W3        = (const float*)d_in[13];
    const float* b3        = (const float*)d_in[14];
    const int*   nts       = (const int*)d_in[15];
    float* out = (float*)d_out;

    qnet_kernel<<<dim3(BB / 16), dim3(512), 0, stream>>>(
        particles, weights, actions, timev, Wi, bi, Wh_rz, Wh_n, bn,
        W1, b1, W2, b2, W3, b3, nts, out);
}
